// SimilarityGAT_45655502356491
// MI455X (gfx1250) — matrix operand order repaired and checked by reading, not compiled, not run on hardware
//
#include <hip/hip_runtime.h>
#include <hip/hip_bf16.h>

typedef __attribute__((ext_vector_type(16))) _Float16 v16h;
typedef __attribute__((ext_vector_type(8)))  _Float16 v8h;
typedef __attribute__((ext_vector_type(8)))  float    v8f;

#define N_NODES 50000
#define N_EDGES 600000
#define N_TOT   (N_EDGES + N_NODES)   // 650000 (edges + self loops)

// ---------------------------------------------------------------- helpers
__device__ __forceinline__ float wave_sum32(float v) {
#pragma unroll
    for (int m = 16; m >= 1; m >>= 1) v += __shfl_xor(v, m, 32);
    return v;
}
__device__ __forceinline__ float leakyf(float x, float s) { return x > 0.f ? x : x * s; }

// monotonic-bits float atomic max (works for mixed signs, init = -inf)
__device__ __forceinline__ void atomicMaxF(float* a, float v) {
    if (v >= 0.f) atomicMax((int*)a, __float_as_int(v));
    else          atomicMin((unsigned int*)a, __float_as_uint(v));
}
__device__ __forceinline__ v16h combine16(v8h lo, v8h hi) {
    v16h a;
#pragma unroll
    for (int i = 0; i < 8; ++i) { a[i] = lo[i]; a[i + 8] = hi[i]; }
    return a;
}

// ---------------------------------------------------------------- utility kernels
__global__ void fill_f32(float* p, long long n, float v) {
    long long i = (long long)blockIdx.x * blockDim.x + threadIdx.x;
    if (i < n) p[i] = v;
}
__global__ void fill_from_ptr(float* p, long long n, const float* v) {
    long long i = (long long)blockIdx.x * blockDim.x + threadIdx.x;
    if (i < n) p[i] = v[0];
}
__global__ void cvt_f16(const float* x, _Float16* y, long long n) {
    long long i = (long long)blockIdx.x * blockDim.x + threadIdx.x;
    if (i < n) y[i] = (_Float16)x[i];
}
// W (K x C row-major f32) -> Wt (C x K f16, i.e. column-major of W)
__global__ void wT_f16(const float* w, _Float16* wt, int K, int C) {
    int i = blockIdx.x * blockDim.x + threadIdx.x;
    if (i < K * C) { int k = i / C, c = i % C; wt[c * K + k] = (_Float16)w[i]; }
}
__global__ void bias_add(float* x, const float* b, int C, long long total) {
    long long i = (long long)blockIdx.x * blockDim.x + threadIdx.x;
    if (i < total) x[i] += b[i % C];
}

// ---------------------------------------------------------------- degree / self-loop attr
__global__ void deg_kernel(const int* dst0, const float* ea, float* s, float* cnt) {
    int e = blockIdx.x * blockDim.x + threadIdx.x;
    if (e < N_EDGES) {
        atomicAdd(&s[dst0[e]], ea[e]);
        atomicAdd(&cnt[dst0[e]], 1.f);
    }
}
__global__ void loop_attr_kernel(const float* s, const float* cnt, float* la) {
    int i = blockIdx.x * blockDim.x + threadIdx.x;
    if (i < N_NODES) la[i] = s[i] / fmaxf(cnt[i], 1.f);
}

// ---------------------------------------------------------------- layer norm (+ optional leaky)
// one wave per row, 4 rows per block; C in {64, 256}; in-place safe
__global__ void ln_kernel(const float* x, const float* g, const float* b, float* out,
                          int C, float slope) {
    int row = blockIdx.x * 4 + (threadIdx.x >> 5);
    int lane = threadIdx.x & 31;
    if (row >= N_NODES) return;
    int per = C >> 5;
    float vals[8];
    float s = 0.f, s2 = 0.f;
    const float* rp = x + (size_t)row * C;
#pragma unroll
    for (int i = 0; i < 8; ++i) {
        if (i < per) {
            float v = rp[lane + i * 32];
            vals[i] = v; s += v; s2 += v * v;
        }
    }
    s = wave_sum32(s); s2 = wave_sum32(s2);
    float mu = s / C;
    float var = s2 / C - mu * mu;
    float rs = rsqrtf(var + 1e-5f);
    float* op = out + (size_t)row * C;
#pragma unroll
    for (int i = 0; i < 8; ++i) {
        if (i < per) {
            int c = lane + i * 32;
            float o = (vals[i] - mu) * rs * g[c] + b[c];
            op[c] = leakyf(o, slope);
        }
    }
}

// ---------------------------------------------------------------- WMMA GEMM: D(MxC) = A(MxK,f16) * Bt(CxK,f16)
// one wave per 16x16 tile; K multiple of 32
__global__ void wmma_gemm(const _Float16* __restrict__ A, const _Float16* __restrict__ Bt,
                          float* __restrict__ D, int K, int C) {
    int mt = blockIdx.x, ct = blockIdx.y;
    int lane = threadIdx.x;
    int half = lane >> 4;       // lane half selects K sub-range per ISA layout
    int l = lane & 15;
    const _Float16* ap = A + (size_t)(mt * 16 + l) * K;
    const _Float16* bp = Bt + (size_t)(ct * 16 + l) * K;
    v8f acc = {};
    for (int k = 0; k < K; k += 32) {
        int ka = k + half * 8;                       // A: V0-3 = K ka..ka+7
        v8h alo = *(const v8h*)(ap + ka);
        v8h ahi = *(const v8h*)(ap + ka + 16);       // A: V4-7 = K ka+16..ka+23
        v16h a = combine16(alo, ahi);
        v16h bm = combine16(*(const v8h*)(bp + ka), *(const v8h*)(bp + ka + 16));// B: same K order as A
        acc = __builtin_amdgcn_wmma_f32_16x16x32_f16(false, a, false, bm,
                                                     (short)0, acc, false, false);
    }
    int col = ct * 16 + l;
    int rbase = mt * 16 + half * 8;                  // D: lanes0-15 rows 0-7, lanes16-31 rows 8-15
#pragma unroll
    for (int r = 0; r < 8; ++r) D[(size_t)(rbase + r) * C + col] = acc[r];
}

// ---------------------------------------------------------------- attention dots a_s, a_d
// one wave per (node, head); feature slice length 64
__global__ void attdot_kernel(const float* xh, const float* att_s, const float* att_d,
                              float* a_s, float* a_d, int H) {
    int wid = blockIdx.x * (blockDim.x >> 5) + (threadIdx.x >> 5);
    int lane = threadIdx.x & 31;
    if (wid >= N_NODES * H) return;
    int n = wid / H, h = wid % H;
    int C = H * 64;
    const float* row = xh + (size_t)n * C + h * 64;
    float v0 = row[lane], v1 = row[lane + 32];
    float ss = v0 * att_s[h * 64 + lane] + v1 * att_s[h * 64 + lane + 32];
    float dd = v0 * att_d[h * 64 + lane] + v1 * att_d[h * 64 + lane + 32];
    ss = wave_sum32(ss); dd = wave_sum32(dd);
    if (lane == 0) { a_s[wid] = ss; a_d[wid] = dd; }
}

// we_dot[h] = sum_c We[0, h*64+c] * att_e[h, c]   (launch with 32*H threads)
__global__ void wedot_kernel(const float* We, const float* att_e, float* out) {
    int h = threadIdx.x >> 5, lane = threadIdx.x & 31;
    float s = We[h * 64 + lane] * att_e[h * 64 + lane]
            + We[h * 64 + lane + 32] * att_e[h * 64 + lane + 32];
    s = wave_sum32(s);
    if (lane == 0) out[h] = s;
}

// ---------------------------------------------------------------- edge softmax (3 passes, alpha in d_out)
__global__ void alpha_raw_kernel(const int* src0, const int* dst0, const float* ea,
                                 const float* lattr, const float* a_s, const float* a_d,
                                 const float* wedot, float* alpha, float* mx, int H) {
    long long idx = (long long)blockIdx.x * blockDim.x + threadIdx.x;
    if (idx >= (long long)N_TOT * H) return;
    int e = (int)(idx / H), h = (int)(idx % H);
    int s = (e < N_EDGES) ? src0[e] : (e - N_EDGES);
    int d = (e < N_EDGES) ? dst0[e] : (e - N_EDGES);
    float eav = (e < N_EDGES) ? ea[e] : lattr[e - N_EDGES];
    float r = leakyf(a_s[s * H + h] + a_d[d * H + h] + eav * wedot[h], 0.2f);
    alpha[idx] = r;
    atomicMaxF(&mx[d * H + h], r);
}
__global__ void alpha_exp_kernel(const int* dst0, float* alpha, const float* mx,
                                 float* den, int H) {
    long long idx = (long long)blockIdx.x * blockDim.x + threadIdx.x;
    if (idx >= (long long)N_TOT * H) return;
    int e = (int)(idx / H), h = (int)(idx % H);
    int d = (e < N_EDGES) ? dst0[e] : (e - N_EDGES);
    float ex = __expf(alpha[idx] - mx[d * H + h]);
    alpha[idx] = ex;
    atomicAdd(&den[d * H + h], ex);
}
__global__ void alpha_norm_kernel(const int* dst0, float* alpha, const float* den, int H) {
    long long idx = (long long)blockIdx.x * blockDim.x + threadIdx.x;
    if (idx >= (long long)N_TOT * H) return;
    int e = (int)(idx / H), h = (int)(idx % H);
    int d = (e < N_EDGES) ? dst0[e] : (e - N_EDGES);
    alpha[idx] /= den[d * H + h];
}

// ---------------------------------------------------------------- scatter aggregation
__global__ void agg_kernel(const int* src0, const int* dst0, const float* xh,
                           const float* alpha, float* agg, int H) {
    int C = H * 64;
    long long total = (long long)N_TOT * C;
    long long idx = (long long)blockIdx.x * blockDim.x + threadIdx.x;
    if (idx >= total) return;
    int e = (int)(idx / C), c = (int)(idx % C), h = c >> 6;
    int s = (e < N_EDGES) ? src0[e] : (e - N_EDGES);
    int d = (e < N_EDGES) ? dst0[e] : (e - N_EDGES);
    atomicAdd(&agg[(size_t)d * C + c], xh[(size_t)s * C + c] * alpha[(long long)e * H + h]);
}

// conv3 epilogue: mean over 4 heads + bias + leaky
__global__ void headmean_kernel(const float* agg, const float* bias, float* out) {
    long long i = (long long)blockIdx.x * blockDim.x + threadIdx.x;
    if (i >= (long long)N_NODES * 64) return;
    int n = (int)(i >> 6), c = (int)(i & 63);
    const float* r = agg + (size_t)n * 256;
    float v = 0.25f * (r[c] + r[64 + c] + r[128 + c] + r[192 + c]);
    out[i] = leakyf(v + bias[c], 0.01f);
}

// ---------------------------------------------------------------- decoder: WMMA on gathered [x[src]|x[dst]] rows
// grid (E/16, 4); K=128; fuses +b1, leaky, dot with W2, atomicAdd into pred
__global__ void decoder_kernel(const _Float16* __restrict__ X, const int* src0, const int* dst0,
                               const _Float16* __restrict__ W1t, const float* b1,
                               const float* w2, float* pred) {
    int et = blockIdx.x, ct = blockIdx.y;
    int lane = threadIdx.x;
    int half = lane >> 4;
    int l = lane & 15;
    int e = et * 16 + l;
    int sn = src0[e], dn = dst0[e];
    const _Float16* wp = W1t + (size_t)(ct * 16 + l) * 128;
    v8f acc = {};
#pragma unroll
    for (int k = 0; k < 128; k += 32) {
        int ka = k + half * 8;
        int kb = ka + 16;
        const _Float16* ra = (ka < 64) ? (X + (size_t)sn * 64 + ka)
                                       : (X + (size_t)dn * 64 + (ka - 64));
        const _Float16* rb = (kb < 64) ? (X + (size_t)sn * 64 + kb)
                                       : (X + (size_t)dn * 64 + (kb - 64));
        v16h a = combine16(*(const v8h*)ra, *(const v8h*)rb);
        v16h bm = combine16(*(const v8h*)(wp + ka), *(const v8h*)(wp + kb));
        acc = __builtin_amdgcn_wmma_f32_16x16x32_f16(false, a, false, bm,
                                                     (short)0, acc, false, false);
    }
    int col = ct * 16 + l;
    float bb = b1[col], ww = w2[col];
#pragma unroll
    for (int r = 0; r < 8; ++r) {
        float v = leakyf(acc[r] + bb, 0.01f) * ww;
#pragma unroll
        for (int m = 8; m >= 1; m >>= 1) v += __shfl_xor(v, m, 16);
        if (l == 0) atomicAdd(&pred[et * 16 + half * 8 + r], v);
    }
}

// ---------------------------------------------------------------- host
static inline unsigned gblk(long long n, int b) { return (unsigned)((n + b - 1) / b); }

extern "C" void kernel_launch(void* const* d_in, const int* in_sizes, int n_in,
                              void* d_out, int out_size, void* d_ws, size_t ws_size,
                              hipStream_t stream) {
    // ---- inputs (setup_inputs dict order)
    const float* emb   = (const float*)d_in[0];
    const float* n1_g  = (const float*)d_in[1];
    const float* n1_b  = (const float*)d_in[2];
    const float* c1_W  = (const float*)d_in[3];
    const float* c1_as = (const float*)d_in[4];
    const float* c1_ad = (const float*)d_in[5];
    const float* c1_We = (const float*)d_in[6];
    const float* c1_ae = (const float*)d_in[7];
    const float* c1_b  = (const float*)d_in[8];
    const float* n2_g  = (const float*)d_in[9];
    const float* n2_b  = (const float*)d_in[10];
    const float* c2_W  = (const float*)d_in[11];
    const float* c2_as = (const float*)d_in[12];
    const float* c2_ad = (const float*)d_in[13];
    const float* c2_We = (const float*)d_in[14];
    const float* c2_ae = (const float*)d_in[15];
    const float* c2_b  = (const float*)d_in[16];
    const float* n3_g  = (const float*)d_in[17];
    const float* n3_b  = (const float*)d_in[18];
    const float* c3_W  = (const float*)d_in[19];
    const float* c3_as = (const float*)d_in[20];
    const float* c3_ad = (const float*)d_in[21];
    const float* c3_We = (const float*)d_in[22];
    const float* c3_ae = (const float*)d_in[23];
    const float* c3_b  = (const float*)d_in[24];
    const float* dW1   = (const float*)d_in[25];
    const float* db1   = (const float*)d_in[26];
    const float* dW2   = (const float*)d_in[27];
    const float* db2   = (const float*)d_in[28];
    const int*   eidx  = (const int*)d_in[29];
    const float* eattr = (const float*)d_in[30];
    const int* src0 = eidx;
    const int* dst0 = eidx + N_EDGES;

    // ---- outputs: pred | a1 | a2 | a3
    float* out  = (float*)d_out;
    float* pred = out;
    float* A1   = out + 600000;
    float* A2   = out + 3200000;
    float* A3   = out + 3850000;

    // ---- workspace layout
    float* B1 = (float*)d_ws;                     // N x 256 f32 (x / agg ping)
    float* B2 = B1 + (size_t)N_NODES * 256;       // N x 256 f32 (xh pong)
    _Float16* XF = (_Float16*)(B2 + (size_t)N_NODES * 256);  // N x 256 f16
    _Float16* WT = XF + (size_t)N_NODES * 256;    // 16384 f16 (transposed weights)
    float* sbuf  = (float*)(WT + 16384);
    float* cnt   = sbuf + 50048;
    float* lattr = cnt + 50048;
    float* a_s   = lattr + 50048;
    float* a_d   = a_s + 200000;
    float* mx    = a_d + 200000;
    float* den   = mx + 200000;
    float* wedot = den + 200000;

    const float NEG_INF = -__builtin_huge_valf();

    // ---- self-loop attr
    fill_f32<<<gblk(N_NODES, 256), 256, 0, stream>>>(sbuf, N_NODES, 0.f);
    fill_f32<<<gblk(N_NODES, 256), 256, 0, stream>>>(cnt, N_NODES, 0.f);
    deg_kernel<<<gblk(N_EDGES, 256), 256, 0, stream>>>(dst0, eattr, sbuf, cnt);
    loop_attr_kernel<<<gblk(N_NODES, 256), 256, 0, stream>>>(sbuf, cnt, lattr);

    // =============== conv1 (H=4, 64 -> 256) ===============
    ln_kernel<<<gblk(N_NODES, 4), 128, 0, stream>>>(emb, n1_g, n1_b, B1, 64, 1.0f);
    cvt_f16<<<gblk((long long)N_NODES * 64, 256), 256, 0, stream>>>(B1, XF, (long long)N_NODES * 64);
    wT_f16<<<gblk(64 * 256, 256), 256, 0, stream>>>(c1_W, WT, 64, 256);
    wmma_gemm<<<dim3(N_NODES / 16, 16), 32, 0, stream>>>(XF, WT, B2, 64, 256);
    attdot_kernel<<<gblk((long long)N_NODES * 4, 8), 256, 0, stream>>>(B2, c1_as, c1_ad, a_s, a_d, 4);
    wedot_kernel<<<1, 128, 0, stream>>>(c1_We, c1_ae, wedot);
    fill_f32<<<gblk((long long)N_NODES * 4, 256), 256, 0, stream>>>(mx, (long long)N_NODES * 4, NEG_INF);
    fill_f32<<<gblk((long long)N_NODES * 4, 256), 256, 0, stream>>>(den, (long long)N_NODES * 4, 0.f);
    alpha_raw_kernel<<<gblk((long long)N_TOT * 4, 256), 256, 0, stream>>>(src0, dst0, eattr, lattr, a_s, a_d, wedot, A1, mx, 4);
    alpha_exp_kernel<<<gblk((long long)N_TOT * 4, 256), 256, 0, stream>>>(dst0, A1, mx, den, 4);
    alpha_norm_kernel<<<gblk((long long)N_TOT * 4, 256), 256, 0, stream>>>(dst0, A1, den, 4);
    fill_f32<<<gblk((long long)N_NODES * 256, 256), 256, 0, stream>>>(B1, (long long)N_NODES * 256, 0.f);
    agg_kernel<<<gblk((long long)N_TOT * 256, 256), 256, 0, stream>>>(src0, dst0, B2, A1, B1, 4);
    bias_add<<<gblk((long long)N_NODES * 256, 256), 256, 0, stream>>>(B1, c1_b, 256, (long long)N_NODES * 256);

    // =============== conv2 (H=1, 256 -> 64) ===============
    ln_kernel<<<gblk(N_NODES, 4), 128, 0, stream>>>(B1, n2_g, n2_b, B1, 256, 0.01f);
    cvt_f16<<<gblk((long long)N_NODES * 256, 256), 256, 0, stream>>>(B1, XF, (long long)N_NODES * 256);
    wT_f16<<<gblk(256 * 64, 256), 256, 0, stream>>>(c2_W, WT, 256, 64);
    wmma_gemm<<<dim3(N_NODES / 16, 4), 32, 0, stream>>>(XF, WT, B2, 256, 64);
    attdot_kernel<<<gblk((long long)N_NODES, 8), 256, 0, stream>>>(B2, c2_as, c2_ad, a_s, a_d, 1);
    wedot_kernel<<<1, 32, 0, stream>>>(c2_We, c2_ae, wedot);
    fill_f32<<<gblk(N_NODES, 256), 256, 0, stream>>>(mx, N_NODES, NEG_INF);
    fill_f32<<<gblk(N_NODES, 256), 256, 0, stream>>>(den, N_NODES, 0.f);
    alpha_raw_kernel<<<gblk(N_TOT, 256), 256, 0, stream>>>(src0, dst0, eattr, lattr, a_s, a_d, wedot, A2, mx, 1);
    alpha_exp_kernel<<<gblk(N_TOT, 256), 256, 0, stream>>>(dst0, A2, mx, den, 1);
    alpha_norm_kernel<<<gblk(N_TOT, 256), 256, 0, stream>>>(dst0, A2, den, 1);
    fill_f32<<<gblk((long long)N_NODES * 64, 256), 256, 0, stream>>>(B1, (long long)N_NODES * 64, 0.f);
    agg_kernel<<<gblk((long long)N_TOT * 64, 256), 256, 0, stream>>>(src0, dst0, B2, A2, B1, 1);
    bias_add<<<gblk((long long)N_NODES * 64, 256), 256, 0, stream>>>(B1, c2_b, 64, (long long)N_NODES * 64);

    // =============== conv3 (H=4, 64 -> 256, mean heads) ===============
    ln_kernel<<<gblk(N_NODES, 4), 128, 0, stream>>>(B1, n3_g, n3_b, B1, 64, 0.01f);
    cvt_f16<<<gblk((long long)N_NODES * 64, 256), 256, 0, stream>>>(B1, XF, (long long)N_NODES * 64);
    wT_f16<<<gblk(64 * 256, 256), 256, 0, stream>>>(c3_W, WT, 64, 256);
    wmma_gemm<<<dim3(N_NODES / 16, 16), 32, 0, stream>>>(XF, WT, B2, 64, 256);
    attdot_kernel<<<gblk((long long)N_NODES * 4, 8), 256, 0, stream>>>(B2, c3_as, c3_ad, a_s, a_d, 4);
    wedot_kernel<<<1, 128, 0, stream>>>(c3_We, c3_ae, wedot);
    fill_f32<<<gblk((long long)N_NODES * 4, 256), 256, 0, stream>>>(mx, (long long)N_NODES * 4, NEG_INF);
    fill_f32<<<gblk((long long)N_NODES * 4, 256), 256, 0, stream>>>(den, (long long)N_NODES * 4, 0.f);
    alpha_raw_kernel<<<gblk((long long)N_TOT * 4, 256), 256, 0, stream>>>(src0, dst0, eattr, lattr, a_s, a_d, wedot, A3, mx, 4);
    alpha_exp_kernel<<<gblk((long long)N_TOT * 4, 256), 256, 0, stream>>>(dst0, A3, mx, den, 4);
    alpha_norm_kernel<<<gblk((long long)N_TOT * 4, 256), 256, 0, stream>>>(dst0, A3, den, 4);
    fill_f32<<<gblk((long long)N_NODES * 256, 256), 256, 0, stream>>>(B1, (long long)N_NODES * 256, 0.f);
    agg_kernel<<<gblk((long long)N_TOT * 256, 256), 256, 0, stream>>>(src0, dst0, B2, A3, B1, 4);
    headmean_kernel<<<gblk((long long)N_NODES * 64, 256), 256, 0, stream>>>(B1, c3_b, B2);

    // =============== decoder ===============
    cvt_f16<<<gblk((long long)N_NODES * 64, 256), 256, 0, stream>>>(B2, XF, (long long)N_NODES * 64);
    wT_f16<<<gblk(128 * 64, 256), 256, 0, stream>>>(dW1, WT, 128, 64);
    fill_from_ptr<<<gblk(N_EDGES, 256), 256, 0, stream>>>(pred, N_EDGES, db2);
    decoder_kernel<<<dim3(N_EDGES / 16, 4), 32, 0, stream>>>(XF, src0, dst0, WT, db1, dW2, pred);
}
